// RQVAE_12292196401313
// MI455X (gfx1250) — matrix operand order repaired and checked by reading, not compiled, not run on hardware
//
#include <hip/hip_runtime.h>
#include <hip/hip_bf16.h>

// ---------------------------------------------------------------------------
// Types for WMMA
// ---------------------------------------------------------------------------
typedef __attribute__((ext_vector_type(16))) __bf16          v16bf;
typedef __attribute__((ext_vector_type(16))) unsigned short  v16us;
typedef __attribute__((ext_vector_type(8)))  float           v8f;

#define RMS_EPS 1e-6f
#define COMMITF 0.25f

// gfx1250 async global->LDS copy (guarded; falls back to vector copy).
// Probe round 2 revealed the builtin's signature:
//   param0: int __attribute__((vector_size(16))) AS(1)*   (global src)
//   param1: ...                                  AS(3)*   (LDS dst)
//   param2/3: int offset, int cpol
#if defined(__has_builtin)
#if __has_builtin(__builtin_amdgcn_global_load_async_to_lds_b128)
#define USE_ASYNC_LDS 1
#endif
#endif
#ifndef USE_ASYNC_LDS
#define USE_ASYNC_LDS 0
#endif

typedef int v4i_vs __attribute__((vector_size(16)));
typedef v4i_vs __attribute__((address_space(1)))* gv4p_t;   // global v4i*
typedef v4i_vs __attribute__((address_space(3)))* lv4p_t;   // LDS v4i*

// ---------------------------------------------------------------------------
// Helpers
// ---------------------------------------------------------------------------
__device__ inline unsigned short f2bf(float f) {
    unsigned u = __float_as_uint(f);
    unsigned r = (u + 0x7FFFu + ((u >> 16) & 1u)) >> 16;   // round-to-nearest-even
    return (unsigned short)r;
}

// order-preserving float -> u32 (monotone for all finite floats)
__device__ inline unsigned f2ord(float f) {
    unsigned u = __float_as_uint(f);
    return (u & 0x80000000u) ? ~u : (u | 0x80000000u);
}

__device__ inline unsigned long long shflxor_u64(unsigned long long v, int off) {
    unsigned lo = (unsigned)v, hi = (unsigned)(v >> 32);
    lo = __shfl_xor(lo, off, 32);
    hi = __shfl_xor(hi, off, 32);
    return ((unsigned long long)hi << 32) | lo;
}

__device__ inline float waveReduceSum(float v) {
    #pragma unroll
    for (int off = 16; off >= 1; off >>= 1) v += __shfl_xor(v, off, 32);
    return v;
}

__device__ inline float blockReduceSum(float v) {
    __shared__ float s[32];
    int lane = threadIdx.x & 31, w = threadIdx.x >> 5;
    v = waveReduceSum(v);
    if (lane == 0) s[w] = v;
    __syncthreads();
    int nw = (blockDim.x + 31) >> 5;
    v = (threadIdx.x < (unsigned)nw) ? s[threadIdx.x] : 0.f;
    if (w == 0) v = waveReduceSum(v);
    return v;   // valid on thread 0
}

// ---------------------------------------------------------------------------
// 1. Input normalization: xn = (x - mean)/std ; also emit bf16 mirror
// ---------------------------------------------------------------------------
__global__ void k_normalize(const float* __restrict__ x, const float* __restrict__ mean,
                            const float* __restrict__ stdv, float* __restrict__ xn,
                            unsigned short* __restrict__ xnb, int total, int D) {
    int i = blockIdx.x * blockDim.x + threadIdx.x;
    if (i < total) {
        int d = i % D;
        float v = (x[i] - mean[d]) / stdv[d];
        xn[i] = v;
        xnb[i] = f2bf(v);
    }
}

// ---------------------------------------------------------------------------
// 2. f32 -> bf16 tensor conversion (weights, once per launch)
// ---------------------------------------------------------------------------
__global__ void k_conv_bf16(const float* __restrict__ src, unsigned short* __restrict__ dst,
                            size_t n) {
    size_t i = (size_t)blockIdx.x * blockDim.x + threadIdx.x;
    if (i < n) dst[i] = f2bf(src[i]);
}

// ---------------------------------------------------------------------------
// 3. Codebook prep: f32 -> bf16 + per-row squared norm. One row (128) / block.
// ---------------------------------------------------------------------------
__global__ void k_cb_prep(const float* __restrict__ cb, unsigned short* __restrict__ cbf,
                          float* __restrict__ e2, int D) {
    int row = blockIdx.x;
    int t = threadIdx.x;                       // blockDim.x == D == 128
    float v = cb[(size_t)row * D + t];
    cbf[(size_t)row * D + t] = f2bf(v);
    float s = blockReduceSum(v * v);
    if (t == 0) e2[row] = s;
}

// ---------------------------------------------------------------------------
// 4. Zero / init kernels
// ---------------------------------------------------------------------------
__global__ void k_zero_f32(float* __restrict__ p, size_t n) {
    size_t i = (size_t)blockIdx.x * blockDim.x + threadIdx.x;
    if (i < n) p[i] = 0.f;
}
__global__ void k_init_keys(unsigned long long* __restrict__ keys, int n) {
    int i = blockIdx.x * blockDim.x + threadIdx.x;
    if (i < n) keys[i] = 0xFFFFFFFFFFFFFFFFull;
}

// ---------------------------------------------------------------------------
// 5. WMMA GEMM (all-bf16 operands): out[M,N] = act(A @ W + bias)
//    Block: 256 threads (8 waves), tile 128(M) x 32(N), K-step 32.
//    Per wave per slab: 1 A fragment, 2 B fragments -> 2 WMMAs.
// ---------------------------------------------------------------------------
__global__ void k_gemm_bf16(const unsigned short* __restrict__ A,  // [M,K] bf16
                            const unsigned short* __restrict__ W,  // [K,N] bf16
                            const float* __restrict__ bias, float* __restrict__ out,
                            int M, int N, int K, int relu) {
    __shared__ unsigned short lA[128 * 32];    // 8 KB
    __shared__ unsigned short lB[32 * 32];     // 2 KB

    const int tid  = threadIdx.x;
    const int lane = tid & 31;
    const int wave = tid >> 5;
    const int m0 = blockIdx.x * 128;
    const int n0 = blockIdx.y * 32;
    const int lm = lane & 15;
    const int lh = lane >> 4;

    v8f c0 = {}, c1 = {};

    for (int k0 = 0; k0 < K; k0 += 32) {
        // ---- stage A slab [128 x 32] bf16: pure 16B vector copies
        {
            int row  = tid >> 1;
            int half = tid & 1;
            const unsigned short* s = A + (size_t)(m0 + row) * K + k0 + half * 16;
            const uint4* src = (const uint4*)s;
            uint4* dst = (uint4*)(lA + row * 32 + half * 16);
            dst[0] = src[0];
            dst[1] = src[1];
            if (k0 + 32 < K)
                __builtin_prefetch((const void*)(s + 32), 0, 0);  // global_prefetch next slab
        }
        // ---- stage B slab transposed: lB[n*32+k] = W[(k0+k)*N + n0+n]
        {
            int n = tid >> 3;
            int k = (tid & 7) * 4;
            unsigned short* d = lB + n * 32 + k;
            const unsigned short* s = W + (size_t)(k0 + k) * N + n0 + n;
            d[0] = s[0];
            d[1] = s[(size_t)N];
            d[2] = s[(size_t)2 * N];
            d[3] = s[(size_t)3 * N];
        }
        __syncthreads();

        // ---- fragments in contiguous k order (k = lh*16 + i) for both A and B
        v16us ua, ub0, ub1;
        const unsigned short* pa = lA + (wave * 16 + lm) * 32;
        #pragma unroll
        for (int i = 0; i < 8; ++i) {
            ua[i]     = pa[lh * 16 + i];           // VGPR0-3: K = lh*16 + i
            ua[8 + i] = pa[lh * 16 + 8 + i];       // VGPR4-7: K = lh*16 + 8 + i
        }
        const unsigned short* pb0 = lB + lm * 32 + lh * 16;
        const unsigned short* pb1 = lB + (16 + lm) * 32 + lh * 16;
        #pragma unroll
        for (int i = 0; i < 16; ++i) { ub0[i] = pb0[i]; ub1[i] = pb1[i]; }

        v16bf a = __builtin_bit_cast(v16bf, ua);
        c0 = __builtin_amdgcn_wmma_f32_16x16x32_bf16(
                 false, a, false, __builtin_bit_cast(v16bf, ub0), (short)0, c0, false, false);
        c1 = __builtin_amdgcn_wmma_f32_16x16x32_bf16(
                 false, a, false, __builtin_bit_cast(v16bf, ub1), (short)0, c1, false, false);
        __syncthreads();
    }

    // ---- epilogue: bias + optional ReLU, store per C/D layout
    const int na = n0 + lm;
    const int nb = n0 + 16 + lm;
    const float bva = bias ? bias[na] : 0.f;
    const float bvb = bias ? bias[nb] : 0.f;
    #pragma unroll
    for (int r = 0; r < 8; ++r) {
        int m = m0 + wave * 16 + lh * 8 + r;
        float v0 = c0[r] + bva;
        float v1 = c1[r] + bvb;
        if (relu) { v0 = fmaxf(v0, 0.f); v1 = fmaxf(v1, 0.f); }
        out[(size_t)m * N + na] = v0;
        out[(size_t)m * N + nb] = v1;
    }
}

// ---------------------------------------------------------------------------
// 6. RMSNorm (in place, one row per block) + bf16 mirror for the next GEMM
// ---------------------------------------------------------------------------
__global__ void k_rmsnorm(float* __restrict__ h, const float* __restrict__ g, int N,
                          unsigned short* __restrict__ out_bf) {
    int row = blockIdx.x;
    float* p = h + (size_t)row * N;
    float ss = 0.f;
    for (int i = threadIdx.x; i < N; i += blockDim.x) { float v = p[i]; ss += v * v; }
    ss = blockReduceSum(ss);
    __shared__ float s_rs;
    if (threadIdx.x == 0) s_rs = rsqrtf(ss / (float)N + RMS_EPS);
    __syncthreads();
    float rs = s_rs;
    for (int i = threadIdx.x; i < N; i += blockDim.x) {
        float v = p[i] * rs * g[i];
        p[i] = v;
        if (out_bf) out_bf[(size_t)row * N + i] = f2bf(v);
    }
}

// ---------------------------------------------------------------------------
// 7. VQ argmin: scores S = R[B,128] @ E^T, fused argmin_k(||e_k||^2 - 2 S).
//    Block: 256 threads, tile 32 rows x 128 codewords; wave w -> 16 codewords,
//    one B fragment reused by two A fragments (2 WMMAs per codebook fetch).
//    Keys packed (ord(dist)<<32 | idx) -> atomicMin(u64) == hard argmin with
//    smallest-index tie-break (matches jnp.argmin).
// ---------------------------------------------------------------------------
__global__ void k_vq_argmin(const unsigned short* __restrict__ Rb,   // [B,128] bf16
                            const unsigned short* __restrict__ Cb,   // [K,128] bf16
                            const float* __restrict__ e2,            // [K]
                            unsigned long long* __restrict__ keys) { // [B]
    __shared__ unsigned short lA[32 * 128];    // 8 KB

    const int tid  = threadIdx.x;
    const int lane = tid & 31;
    const int wave = tid >> 5;
    const int m0 = blockIdx.x * 32;
    const int n0 = blockIdx.y * 128 + wave * 16;
    const int lm = lane & 15;
    const int lh = lane >> 4;

    // stage the 32x128 residual tile (shared by all 8 waves): 32B per thread
    {
        int base = tid * 16;                     // 4096 elems / 256 threads
        int row = base >> 7, col = base & 127;
        const unsigned short* src = Rb + (size_t)(m0 + row) * 128 + col;
        unsigned short* dst = lA + row * 128 + col;
#if USE_ASYNC_LDS
        gv4p_t g = (gv4p_t)(uintptr_t)(const void*)src;
        lv4p_t l = (lv4p_t)(uintptr_t)(void*)dst;
        __builtin_amdgcn_global_load_async_to_lds_b128(g, l, 0, 0);
        __builtin_amdgcn_global_load_async_to_lds_b128(g, l, 16, 0);
#if __has_builtin(__builtin_amdgcn_s_wait_asynccnt)
        __builtin_amdgcn_s_wait_asynccnt(0);
#else
        asm volatile("s_wait_asynccnt 0x0" ::: "memory");
#endif
#else
        const uint4* s4 = (const uint4*)src;
        uint4* d4 = (uint4*)dst;
        d4[0] = s4[0];
        d4[1] = s4[1];
#endif
    }
    __syncthreads();

    v8f c0 = {}, c1 = {};
    const unsigned short* prowA0 = lA + lm * 128;
    const unsigned short* prowA1 = lA + (16 + lm) * 128;
    const unsigned short* prowB  = Cb + (size_t)(n0 + lm) * 128;   // E rows == B^T frags
    #pragma unroll
    for (int kk = 0; kk < 128; kk += 32) {
        v16us ua0, ua1, ub;
        #pragma unroll
        for (int i = 0; i < 8; ++i) {
            ua0[i]     = prowA0[kk + lh * 8 + i];
            ua0[8 + i] = prowA0[kk + 16 + lh * 8 + i];
            ua1[i]     = prowA1[kk + lh * 8 + i];
            ua1[8 + i] = prowA1[kk + 16 + lh * 8 + i];
        }
        #pragma unroll
        for (int i = 0; i < 16; ++i) ub[i] = prowB[kk + (i < 8 ? lh * 8 + i : 8 + lh * 8 + i)];
        v16bf b = __builtin_bit_cast(v16bf, ub);
        c0 = __builtin_amdgcn_wmma_f32_16x16x32_bf16(
                 false, __builtin_bit_cast(v16bf, ua0), false, b, (short)0, c0, false, false);
        c1 = __builtin_amdgcn_wmma_f32_16x16x32_bf16(
                 false, __builtin_bit_cast(v16bf, ua1), false, b, (short)0, c1, false, false);
    }

    const int n = n0 + lm;
    const float en = e2[n];
    v8f cc[2] = { c0, c1 };
    #pragma unroll
    for (int s = 0; s < 2; ++s) {
        #pragma unroll
        for (int r = 0; r < 8; ++r) {
            float val = en - 2.0f * cc[s][r];
            unsigned long long key = ((unsigned long long)f2ord(val) << 32) | (unsigned)n;
            #pragma unroll
            for (int off = 8; off >= 1; off >>= 1) {   // min across the 16-lane half
                unsigned long long o = shflxor_u64(key, off);
                if (o < key) key = o;
            }
            if (lm == 0) {
                int m = m0 + s * 16 + lh * 8 + r;
                atomicMin(&keys[m], key);
            }
        }
    }
}

// ---------------------------------------------------------------------------
// 8. VQ update: gather winner (f32), update residual/qsum, loss, ids.
//    One row per block, 128 threads. On the last quantizer also emit qsum bf16.
// ---------------------------------------------------------------------------
__global__ void k_vq_update(const unsigned long long* __restrict__ keys,
                            const float* __restrict__ cb,          // [K,128] f32
                            float* __restrict__ residual,          // [B,128]
                            unsigned short* __restrict__ rbf,      // [B,128] bf16
                            float* __restrict__ qsum,              // [B,128]
                            unsigned short* __restrict__ qs_bf,    // [B,128] bf16 or null
                            float* __restrict__ ids_out,           // [B,4] as float
                            float* __restrict__ vq_acc, int q) {
    int row = blockIdx.x;
    int t = threadIdx.x;
    unsigned idx = (unsigned)(keys[row] & 0xFFFFFFFFull);
    float e = cb[(size_t)idx * 128 + t];
    size_t off = (size_t)row * 128 + t;
    float r = residual[off];
    float d = r - e;
    float qs = qsum[off] + e;      // straight-through: forward value is just e
    qsum[off] = qs;
    residual[off] = d;
    rbf[off] = f2bf(d);
    if (qs_bf) qs_bf[off] = f2bf(qs);
    float s = blockReduceSum(d * d);
    if (t == 0) {
        atomicAdd(vq_acc, s);
        ids_out[(size_t)row * 4 + q] = (float)idx;
    }
}

// ---------------------------------------------------------------------------
// 9. Reconstruction loss partial sum + finalize
// ---------------------------------------------------------------------------
__global__ void k_recon_loss(const float* __restrict__ recon, const float* __restrict__ xn,
                             float* __restrict__ acc, size_t n) {
    float ss = 0.f;
    for (size_t i = (size_t)blockIdx.x * blockDim.x + threadIdx.x; i < n;
         i += (size_t)gridDim.x * blockDim.x) {
        float v = recon[i] - xn[i];
        ss += v * v;
    }
    ss = blockReduceSum(ss);
    if (threadIdx.x == 0) atomicAdd(acc, ss);
}

__global__ void k_finalize(const float* __restrict__ acc, float* __restrict__ losses,
                           float inv_recon, float vq_scale) {
    if (threadIdx.x == 0 && blockIdx.x == 0) {
        losses[0] = acc[0] * inv_recon;                // recon mse
        losses[1] = acc[1] * vq_scale;                 // (1 + COMMIT) * sum mse_q
    }
}

// ---------------------------------------------------------------------------
// Host launcher
// ---------------------------------------------------------------------------
extern "C" void kernel_launch(void* const* d_in, const int* in_sizes, int n_in,
                              void* d_out, int out_size, void* d_ws, size_t ws_size,
                              hipStream_t stream) {
    (void)in_sizes; (void)n_in; (void)out_size; (void)ws_size;

    const int B = 16384, DIN = 768;
    const int N0 = 512, N1 = 256, N2 = 128;
    const int Q = 4, KCB = 8192;

    const float* x        = (const float*)d_in[0];
    const float* emb_mean = (const float*)d_in[1];
    const float* emb_std  = (const float*)d_in[2];
    const float* enc_w0 = (const float*)d_in[3];
    const float* enc_b0 = (const float*)d_in[4];
    const float* enc_g0 = (const float*)d_in[5];
    const float* enc_w1 = (const float*)d_in[6];
    const float* enc_b1 = (const float*)d_in[7];
    const float* enc_g1 = (const float*)d_in[8];
    const float* enc_w2 = (const float*)d_in[9];
    const float* enc_b2 = (const float*)d_in[10];
    const float* enc_g2 = (const float*)d_in[11];
    const float* codebooks = (const float*)d_in[12];
    const float* dec_w0 = (const float*)d_in[13];
    const float* dec_b0 = (const float*)d_in[14];
    const float* dec_g0 = (const float*)d_in[15];
    const float* dec_w1 = (const float*)d_in[16];
    const float* dec_b1 = (const float*)d_in[17];
    const float* dec_g1 = (const float*)d_in[18];
    const float* dec_w2 = (const float*)d_in[19];
    const float* dec_b2 = (const float*)d_in[20];

    // ---- workspace carve-up (256B aligned)
    char* ws = (char*)d_ws;
    size_t off = 0;
    auto alloc = [&](size_t bytes) -> char* {
        char* p = ws + off;
        off = (off + bytes + 255) & ~(size_t)255;
        return p;
    };
    float* XN  = (float*)alloc((size_t)B * DIN * 4);
    unsigned short* XNb = (unsigned short*)alloc((size_t)B * DIN * 2);
    float* Ha  = (float*)alloc((size_t)B * N0 * 4);   // h0, later y1
    unsigned short* HaB = (unsigned short*)alloc((size_t)B * N0 * 2);
    float* Hb  = (float*)alloc((size_t)B * N1 * 4);   // h1, later y0
    unsigned short* HbB = (unsigned short*)alloc((size_t)B * N1 * 2);
    float* RES = (float*)alloc((size_t)B * N2 * 4);   // h2 / residual
    unsigned short* RBF = (unsigned short*)alloc((size_t)B * N2 * 2);
    float* QS  = (float*)alloc((size_t)B * N2 * 4);   // quantized sum
    unsigned short* QSb = (unsigned short*)alloc((size_t)B * N2 * 2);
    unsigned short* CBF = (unsigned short*)alloc((size_t)Q * KCB * N2 * 2);
    float* E2  = (float*)alloc((size_t)Q * KCB * 4);
    unsigned long long* KEYS = (unsigned long long*)alloc((size_t)B * 8);
    unsigned short* W0B = (unsigned short*)alloc((size_t)DIN * N0 * 2);
    unsigned short* W1B = (unsigned short*)alloc((size_t)N0 * N1 * 2);
    unsigned short* W2B = (unsigned short*)alloc((size_t)N1 * N2 * 2);
    unsigned short* D0B = (unsigned short*)alloc((size_t)N2 * N1 * 2);
    unsigned short* D1B = (unsigned short*)alloc((size_t)N1 * N0 * 2);
    unsigned short* D2B = (unsigned short*)alloc((size_t)N0 * DIN * 2);
    float* ACC = (float*)alloc(256);                   // [0]=recon sum, [1]=vq sum

    float* recon  = (float*)d_out;
    float* ids    = recon + (size_t)B * DIN;
    float* losses = ids + (size_t)B * Q;

    // ---- 1. normalize (+ bf16 mirror)
    {
        int total = B * DIN;
        k_normalize<<<(total + 255) / 256, 256, 0, stream>>>(x, emb_mean, emb_std, XN, XNb, total, DIN);
    }
    // ---- 2. weight conversions (once per launch)
    auto conv = [&](const float* s, unsigned short* d, size_t n) {
        k_conv_bf16<<<(unsigned)((n + 255) / 256), 256, 0, stream>>>(s, d, n);
    };
    conv(enc_w0, W0B, (size_t)DIN * N0);
    conv(enc_w1, W1B, (size_t)N0 * N1);
    conv(enc_w2, W2B, (size_t)N1 * N2);
    conv(dec_w0, D0B, (size_t)N2 * N1);
    conv(dec_w1, D1B, (size_t)N1 * N0);
    conv(dec_w2, D2B, (size_t)N0 * DIN);
    // ---- 3. codebook prep (bf16 + ||e||^2), accumulator zeroing
    k_cb_prep<<<Q * KCB, 128, 0, stream>>>(codebooks, CBF, E2, N2);
    k_zero_f32<<<(unsigned)(((size_t)B * N2 + 255) / 256), 256, 0, stream>>>(QS, (size_t)B * N2);
    k_zero_f32<<<1, 64, 0, stream>>>(ACC, 2);

    // ---- 4. encoder
    k_gemm_bf16<<<dim3(B / 128, N0 / 32), 256, 0, stream>>>(XNb, W0B, enc_b0, Ha, B, N0, DIN, 1);
    k_rmsnorm<<<B, 256, 0, stream>>>(Ha, enc_g0, N0, HaB);
    k_gemm_bf16<<<dim3(B / 128, N1 / 32), 256, 0, stream>>>(HaB, W1B, enc_b1, Hb, B, N1, N0, 1);
    k_rmsnorm<<<B, 256, 0, stream>>>(Hb, enc_g1, N1, HbB);
    k_gemm_bf16<<<dim3(B / 128, N2 / 32), 256, 0, stream>>>(HbB, W2B, enc_b2, RES, B, N2, N1, 1);
    k_rmsnorm<<<B, 128, 0, stream>>>(RES, enc_g2, N2, RBF);   // bf16 residual for VQ

    // ---- 5. residual vector quantization
    for (int q = 0; q < Q; ++q) {
        k_init_keys<<<(B + 255) / 256, 256, 0, stream>>>(KEYS, B);
        k_vq_argmin<<<dim3(B / 32, KCB / 128), 256, 0, stream>>>(
            RBF, CBF + (size_t)q * KCB * N2, E2 + (size_t)q * KCB, KEYS);
        k_vq_update<<<B, 128, 0, stream>>>(
            KEYS, codebooks + (size_t)q * KCB * N2, RES, RBF, QS,
            (q == Q - 1) ? QSb : nullptr, ids, ACC + 1, q);
    }

    // ---- 6. decoder
    k_gemm_bf16<<<dim3(B / 128, N1 / 32), 256, 0, stream>>>(QSb, D0B, dec_b0, Hb, B, N1, N2, 1);
    k_rmsnorm<<<B, 256, 0, stream>>>(Hb, dec_g0, N1, HbB);
    k_gemm_bf16<<<dim3(B / 128, N0 / 32), 256, 0, stream>>>(HbB, D1B, dec_b1, Ha, B, N0, N1, 1);
    k_rmsnorm<<<B, 256, 0, stream>>>(Ha, dec_g1, N0, HaB);
    k_gemm_bf16<<<dim3(B / 128, DIN / 32), 256, 0, stream>>>(HaB, D2B, dec_b2, recon, B, DIN, N0, 0);

    // ---- 7. losses
    k_recon_loss<<<2048, 256, 0, stream>>>(recon, XN, ACC, (size_t)B * DIN);
    k_finalize<<<1, 64, 0, stream>>>(ACC, losses,
                                     1.0f / (float)((size_t)B * DIN),
                                     (1.0f + COMMITF) / (float)((size_t)B * N2));
}
